// AudioMAEConditionQuantResEncoder_77678778515970
// MI455X (gfx1250) — matrix operand order repaired and checked by reading, not compiled, not run on hardware
//
#include <hip/hip_runtime.h>
#include <stdint.h>

// Problem constants
#define NB    8
#define T1    513
#define NTOK  4096          // 8 * 512 tokens
#define DDIM  768           // feature dim (K of the GEMM)
#define KCENT 8192          // number of centroids (N of the GEMM)

// GEMM tiling
#define TM      128         // M rows per workgroup
#define NCHUNK  512         // centroids per workgroup (blockIdx.y)
#define NSUBT   64          // N sub-tile processed per accumulation pass
#define KSTEP   32          // K per WMMA step
#define NKB     (DDIM / KSTEP)   // 24 K-blocks
#define NCHUNKS 16          // KCENT / NCHUNK
#define LDA     40          // padded LDS row stride (uint16), 80B rows (16B aligned)

typedef __attribute__((ext_vector_type(16))) __bf16 v16bf;
typedef __attribute__((ext_vector_type(2)))  __bf16 v2bf;
typedef __attribute__((ext_vector_type(8)))  float  v8f;

// ---------- helpers ----------
static __device__ __forceinline__ uint16_t bf16_rn(float f) {
    uint32_t u = __float_as_uint(f);
    u += 0x7fffu + ((u >> 16) & 1u);   // round-to-nearest-even
    return (uint16_t)(u >> 16);
}
static __device__ __forceinline__ void split_bf16(float x, uint16_t& h, uint16_t& l) {
    uint16_t hb = bf16_rn(x);
    float hf = __uint_as_float(((uint32_t)hb) << 16);
    h = hb;
    l = bf16_rn(x - hf);
}

// ---------- pass 1a: rep (drop t=0) -> hi/lo bf16 planes ----------
__global__ void vq_conv_rep(const float* __restrict__ rep,
                            uint16_t* __restrict__ repHi,
                            uint16_t* __restrict__ repLo) {
    int i = blockIdx.x;                 // token 0..4095
    int b = i >> 9, t = i & 511;
    const float* src = rep + ((size_t)(b * T1 + t + 1)) * DDIM;
    uint16_t* dh = repHi + (size_t)i * DDIM;
    uint16_t* dl = repLo + (size_t)i * DDIM;
    for (int j = threadIdx.x; j < DDIM; j += 256) {
        uint16_t h, l; split_bf16(src[j], h, l);
        dh[j] = h; dl[j] = l;
    }
}

// ---------- pass 1b: centroids -> hi/lo planes + norms ----------
__global__ void vq_conv_cent(const float* __restrict__ cent,
                             uint16_t* __restrict__ centHi,
                             uint16_t* __restrict__ centLo,
                             float* __restrict__ centNorm) {
    int k = blockIdx.x;                 // centroid 0..8191
    const float* src = cent + (size_t)k * DDIM;
    uint16_t* dh = centHi + (size_t)k * DDIM;
    uint16_t* dl = centLo + (size_t)k * DDIM;
    float s = 0.f;
    for (int j = threadIdx.x; j < DDIM; j += 256) {
        float x = src[j];
        uint16_t h, l; split_bf16(x, h, l);
        dh[j] = h; dl[j] = l;
        s += x * x;
    }
    for (int off = 16; off > 0; off >>= 1) s += __shfl_xor(s, off, 32);
    __shared__ float red[8];
    int lane = threadIdx.x & 31, wid = threadIdx.x >> 5;
    if (lane == 0) red[wid] = s;
    __syncthreads();
    if (threadIdx.x == 0) {
        float tt = 0.f;
        for (int q = 0; q < 8; q++) tt += red[q];
        centNorm[k] = tt;
    }
}

// Issue one K-step staging batch: exactly 6 async ops per wave (in-order ASYNCcnt).
static __device__ __forceinline__ void stage_async(
        int tid, int mbase, int nrowbase, int kb,
        uint32_t ldsAh, uint32_t ldsAl, uint32_t ldsBh, uint32_t ldsBl,
        uint64_t gAh, uint64_t gAl, uint64_t gBh, uint64_t gBl) {
#pragma unroll
    for (int i = 0; i < 2; i++) {
        int e = tid + i * 256;          // 0..511 -> 16B chunk of A tile (128x32 bf16)
        int r = e >> 2, c = e & 3;
        uint32_t lo = (uint32_t)(r * (LDA * 2) + c * 16);
        uint32_t go = (uint32_t)((mbase + r) * (DDIM * 2) + kb * 64 + c * 16);
        asm volatile("global_load_async_to_lds_b128 %0, %1, %2"
                     :: "v"(ldsAh + lo), "v"(go), "s"(gAh) : "memory");
        asm volatile("global_load_async_to_lds_b128 %0, %1, %2"
                     :: "v"(ldsAl + lo), "v"(go), "s"(gAl) : "memory");
    }
    {
        int r = tid >> 2, c = tid & 3;  // 256 chunks of B tile (64x32 bf16)
        uint32_t lo = (uint32_t)(r * (LDA * 2) + c * 16);
        uint32_t go = (uint32_t)((nrowbase + r) * (DDIM * 2) + kb * 64 + c * 16);
        asm volatile("global_load_async_to_lds_b128 %0, %1, %2"
                     :: "v"(ldsBh + lo), "v"(go), "s"(gBh) : "memory");
        asm volatile("global_load_async_to_lds_b128 %0, %1, %2"
                     :: "v"(ldsBl + lo), "v"(go), "s"(gBl) : "memory");
    }
}

// ---------- pass 2: 3xBF16 WMMA GEMM + per-chunk argmin, double-buffered DMA ----------
__global__ __launch_bounds__(256)
void vq_gemm(const uint16_t* __restrict__ repHi, const uint16_t* __restrict__ repLo,
             const uint16_t* __restrict__ centHi, const uint16_t* __restrict__ centLo,
             const float* __restrict__ centNorm,
             float* __restrict__ pVal, int* __restrict__ pIdx) {
    __shared__ uint16_t lAh[2][TM * LDA];
    __shared__ uint16_t lAl[2][TM * LDA];
    __shared__ uint16_t lBh[2][NSUBT * LDA];
    __shared__ uint16_t lBl[2][NSUBT * LDA];

    const int tid  = threadIdx.x;
    const int lane = tid & 31;
    const int w    = tid >> 5;          // wave id 0..7 -> owns M strip w*16..w*16+15
    const int half = lane >> 4;
    const int lm   = lane & 15;

    const int mbase = blockIdx.x * TM;
    const int nbase = blockIdx.y * NCHUNK;

    // 64-bit SGPR bases for async GVS addressing
    const uint64_t gAh = (uint64_t)(uintptr_t)repHi;
    const uint64_t gAl = (uint64_t)(uintptr_t)repLo;
    const uint64_t gBh = (uint64_t)(uintptr_t)centHi;
    const uint64_t gBl = (uint64_t)(uintptr_t)centLo;
    // LDS byte offsets (generic addr low 32 bits == DS offset per ISA flat rules)
    const uint32_t ldsAh = (uint32_t)(uintptr_t)&lAh[0][0];
    const uint32_t ldsAl = (uint32_t)(uintptr_t)&lAl[0][0];
    const uint32_t ldsBh = (uint32_t)(uintptr_t)&lBh[0][0];
    const uint32_t ldsBl = (uint32_t)(uintptr_t)&lBl[0][0];
    const uint32_t strA = TM * LDA * 2;      // bytes per A buffer
    const uint32_t strB = NSUBT * LDA * 2;   // bytes per B buffer

    float runV[8];
    int   runI[8];
#pragma unroll
    for (int r = 0; r < 8; r++) { runV[r] = 3.0e38f; runI[r] = 0x7fffffff; }

    for (int ns = 0; ns < NCHUNK / NSUBT; ns++) {
        const int nrowbase = nbase + ns * NSUBT;
        v8f acc[4] = {};                // 4 N-tiles of 16, fp32 accum

        // prologue: stage kb=0 into buffer 0
        stage_async(tid, mbase, nrowbase, 0, ldsAh, ldsAl, ldsBh, ldsBl,
                    gAh, gAl, gBh, gBl);

        for (int kb = 0; kb < NKB; kb++) {
            const int cur = kb & 1;
            // prefetch next K-step into the other buffer (reads of it finished
            // before the previous iteration's trailing barrier)
            if (kb + 1 < NKB) {
                const int nxt = cur ^ 1;
                stage_async(tid, mbase, nrowbase, kb + 1,
                            ldsAh + nxt * strA, ldsAl + nxt * strA,
                            ldsBh + nxt * strB, ldsBl + nxt * strB,
                            gAh, gAl, gBh, gBl);
                // drain only the older 6-op batch; keep the prefetch in flight
                asm volatile("s_wait_asynccnt 0x6" ::: "memory");
            } else {
                asm volatile("s_wait_asynccnt 0x0" ::: "memory");
            }
            __syncthreads();

            const uint32_t* sAh = (const uint32_t*)lAh[cur];
            const uint32_t* sAl = (const uint32_t*)lAl[cur];
            const uint32_t* sBh = (const uint32_t*)lBh[cur];
            const uint32_t* sBl = (const uint32_t*)lBl[cur];

            // ---- A fragments (16x32 bf16, ISA layout) ----
            v16bf aH, aL;
            const int arow = w * 16 + lm;
#pragma unroll
            for (int j = 0; j < 8; j++) {
                int k0 = (j < 4) ? (half * 8 + 2 * j) : (16 + half * 8 + 2 * (j - 4));
                int au = arow * (LDA / 2) + (k0 >> 1);
                v2bf ph = __builtin_bit_cast(v2bf, sAh[au]);
                aH[2 * j] = ph[0]; aH[2 * j + 1] = ph[1];
                v2bf pl = __builtin_bit_cast(v2bf, sAl[au]);
                aL[2 * j] = pl[0]; aL[2 * j + 1] = pl[1];
            }
            // ---- all B fragments ----
            v16bf bH[4], bL[4];
#pragma unroll
            for (int t = 0; t < 4; t++) {
#pragma unroll
                for (int j = 0; j < 8; j++) {
                    int k0 = (j < 4) ? (half * 8 + 2 * j) : (16 + half * 8 + 2 * (j - 4));
                    int bu = (t * 16 + lm) * (LDA / 2) + (k0 >> 1);
                    v2bf ph = __builtin_bit_cast(v2bf, sBh[bu]);
                    bH[t][2 * j] = ph[0]; bH[t][2 * j + 1] = ph[1];
                    v2bf pl = __builtin_bit_cast(v2bf, sBl[bu]);
                    bL[t][2 * j] = pl[0]; bL[t][2 * j + 1] = pl[1];
                }
            }
            // ---- tile-interleaved WMMAs: consecutive ops hit different accums ----
#pragma unroll
            for (int t = 0; t < 4; t++)
                acc[t] = __builtin_amdgcn_wmma_f32_16x16x32_bf16(
                    false, aH, false, bH[t], (short)0, acc[t], false, false);
#pragma unroll
            for (int t = 0; t < 4; t++)
                acc[t] = __builtin_amdgcn_wmma_f32_16x16x32_bf16(
                    false, aH, false, bL[t], (short)0, acc[t], false, false);
#pragma unroll
            for (int t = 0; t < 4; t++)
                acc[t] = __builtin_amdgcn_wmma_f32_16x16x32_bf16(
                    false, aL, false, bH[t], (short)0, acc[t], false, false);
            __syncthreads();
        }

        // ---- score + argmin over this 64-wide N sub-tile ----
        // C layout: VGPR r holds M = half*8 + r, N = lm (per N-tile t).
#pragma unroll
        for (int r = 0; r < 8; r++) {
            float bv = 3.0e38f; int bi = 0x7fffffff;
#pragma unroll
            for (int t = 0; t < 4; t++) {
                int ng = nrowbase + t * 16 + lm;
                float s = centNorm[ng] - 2.0f * acc[t][r];
                if (s < bv || (s == bv && ng < bi)) { bv = s; bi = ng; }
            }
            // reduce over the 16 lanes of this half (masks < 16 stay in-half)
#pragma unroll
            for (int off = 1; off < 16; off <<= 1) {
                float ov = __shfl_xor(bv, off, 32);
                int   oi = __shfl_xor(bi, off, 32);
                if (ov < bv || (ov == bv && oi < bi)) { bv = ov; bi = oi; }
            }
            if (bv < runV[r] || (bv == runV[r] && bi < runI[r])) { runV[r] = bv; runI[r] = bi; }
        }
    }

    // one lane per half writes the 8 rows it owns
    if (lm == 0) {
        int mrow = mbase + w * 16 + half * 8;
#pragma unroll
        for (int r = 0; r < 8; r++) {
            pVal[(size_t)(mrow + r) * NCHUNKS + blockIdx.y] = runV[r];
            pIdx[(size_t)(mrow + r) * NCHUNKS + blockIdx.y] = runI[r];
        }
    }
}

// ---------- pass 3: reduce chunk partials, emit tokens + gather ----------
__global__ void vq_reduce_gather(const float* __restrict__ pVal,
                                 const int* __restrict__ pIdx,
                                 const float* __restrict__ cent,
                                 int* __restrict__ outTok,
                                 float* __restrict__ outQ) {
    int i = blockIdx.x;                 // token
    __shared__ int sBest;
    if (threadIdx.x == 0) {
        float bv = pVal[(size_t)i * NCHUNKS];
        int   bi = pIdx[(size_t)i * NCHUNKS];
        for (int c = 1; c < NCHUNKS; c++) {
            float v = pVal[(size_t)i * NCHUNKS + c];
            int   d = pIdx[(size_t)i * NCHUNKS + c];
            if (v < bv || (v == bv && d < bi)) { bv = v; bi = d; }
        }
        outTok[i] = bi;
        sBest = bi;
    }
    __syncthreads();
    const float* src = cent + (size_t)sBest * DDIM;
    float* dst = outQ + (size_t)i * DDIM;
    for (int j = threadIdx.x; j < DDIM; j += 256) dst[j] = src[j];
}

// ---------- host launcher ----------
extern "C" void kernel_launch(void* const* d_in, const int* in_sizes, int n_in,
                              void* d_out, int out_size, void* d_ws, size_t ws_size,
                              hipStream_t stream) {
    (void)in_sizes; (void)n_in; (void)out_size; (void)ws_size;
    const float* rep  = (const float*)d_in[0];
    const float* cent = (const float*)d_in[1];

    char* ws = (char*)d_ws;
    size_t off = 0;
    uint16_t* repHi  = (uint16_t*)(ws + off); off += (size_t)NTOK  * DDIM * 2;
    uint16_t* repLo  = (uint16_t*)(ws + off); off += (size_t)NTOK  * DDIM * 2;
    uint16_t* centHi = (uint16_t*)(ws + off); off += (size_t)KCENT * DDIM * 2;
    uint16_t* centLo = (uint16_t*)(ws + off); off += (size_t)KCENT * DDIM * 2;
    float*    centNorm = (float*)(ws + off);  off += (size_t)KCENT * 4;
    float*    pVal   = (float*)(ws + off);    off += (size_t)NTOK * NCHUNKS * 4;
    int*      pIdx   = (int*)(ws + off);      off += (size_t)NTOK * NCHUNKS * 4;

    int*   outTok = (int*)d_out;                 // tokens as int32 bit patterns
    float* outQ   = (float*)d_out + NTOK;        // quantized vectors follow

    vq_conv_rep<<<NTOK, 256, 0, stream>>>(rep, repHi, repLo);
    vq_conv_cent<<<KCENT, 256, 0, stream>>>(cent, centHi, centLo, centNorm);
    vq_gemm<<<dim3(NTOK / TM, NCHUNKS), 256, 0, stream>>>(
        repHi, repLo, centHi, centLo, centNorm, pVal, pIdx);
    vq_reduce_gather<<<NTOK, 256, 0, stream>>>(pVal, pIdx, cent, outTok, outQ);
}
